// BinaryConv2D_77223511982717
// MI455X (gfx1250) — 8-bit matrix operand order repaired and checked by reading, compiled, not run on hardware
//
#include <hip/hip_runtime.h>

typedef __attribute__((ext_vector_type(8))) int v8i;

#define CIN   64
#define COUT  64
#define IMG_W 128
#define IMG_H 128
#define TILE_W 130

__global__ __launch_bounds__(256) void binarize_weights(
    const float* __restrict__ w, signed char* __restrict__ wb) {
  int idx = blockIdx.x * 256 + threadIdx.x;
  if (idx >= 9 * 64 * 64) return;
  int co  = idx & 63;
  int ci  = (idx >> 6) & 63;
  int tap = idx >> 12;
  float v = w[((size_t)tap * 64 + ci) * 64 + co];
  signed char s = (v > 0.f) ? 1 : ((v < 0.f) ? -1 : 0);
  wb[tap * 4096 + co * 64 + ci] = s;
}

__global__ __launch_bounds__(256) void bconv_wmma(
    const float* __restrict__ x, const signed char* __restrict__ wb,
    float* __restrict__ out) {
  __shared__ alignas(16) signed char xs[3 * TILE_W * CIN];
  __shared__ alignas(16) signed char wsh[9 * COUT * CIN];

  const int nh  = blockIdx.x;
  const int h   = nh & (IMG_H - 1);
  const int tid = threadIdx.x;

  {
    const int4* src = (const int4*)wb;
    int4* dst = (int4*)wsh;
    #pragma unroll
    for (int i = 0; i < 9; ++i) dst[tid + i * 256] = src[tid + i * 256];
  }

  {
    uint32_t* dst = (uint32_t*)xs;
    const int ndw = 3 * TILE_W * CIN / 4;
    for (int i = tid; i < ndw; i += 256) {
      int c4  = i & 15;
      int pos = (i >> 4) % TILE_W;
      int row = (i >> 4) / TILE_W;
      int ih  = h + row - 1;
      int iw  = pos - 1;
      uint32_t packed = 0;
      if ((unsigned)ih < (unsigned)IMG_H && (unsigned)iw < (unsigned)IMG_W) {
        const float4 v = *(const float4*)(
            x + (((size_t)(nh + row - 1)) * IMG_W + iw) * CIN + c4 * 4);
        uint32_t b0 = (uint32_t)(uint8_t)(v.x > 0.f ? 1 : (v.x < 0.f ? -1 : 0));
        uint32_t b1 = (uint32_t)(uint8_t)(v.y > 0.f ? 1 : (v.y < 0.f ? -1 : 0));
        uint32_t b2 = (uint32_t)(uint8_t)(v.z > 0.f ? 1 : (v.z < 0.f ? -1 : 0));
        uint32_t b3 = (uint32_t)(uint8_t)(v.w > 0.f ? 1 : (v.w < 0.f ? -1 : 0));
        packed = b0 | (b1 << 8) | (b2 << 16) | (b3 << 24);
      }
      dst[i] = packed;
    }
  }
  __syncthreads();

  const int lane = tid & 31;
  const int wv   = tid >> 5;
  const int m    = lane & 15;
  const int hi   = lane >> 4;

  v8i acc[4];
  #pragma unroll
  for (int nt = 0; nt < 4; ++nt) acc[nt] = (v8i){0, 0, 0, 0, 0, 0, 0, 0};

  #pragma unroll
  for (int tap = 0; tap < 9; ++tap) {
    const int kh = tap / 3, kw = tap % 3;

    const int rowbase = (kh * TILE_W + wv * 16 + m + kw) * CIN;
    uint64_t a0 = *(const uint64_t*)(xs + rowbase + 16 * hi);
    uint64_t a1 = *(const uint64_t*)(xs + rowbase + 8 + 16 * hi);
    uint64_t a2 = *(const uint64_t*)(xs + rowbase + 32 + 16 * hi);
    uint64_t a3 = *(const uint64_t*)(xs + rowbase + 40 + 16 * hi);
    v8i a;
    a[0] = (int)a0; a[1] = (int)(a0 >> 32);
    a[2] = (int)a1; a[3] = (int)(a1 >> 32);
    a[4] = (int)a2; a[5] = (int)(a2 >> 32);
    a[6] = (int)a3; a[7] = (int)(a3 >> 32);

    const signed char* wt = wsh + tap * 4096;
    #pragma unroll
    for (int nt = 0; nt < 4; ++nt) {
      const signed char* col = wt + (nt * 16 + m) * CIN;
      int4 blo = *(const int4*)(col + 16 * hi);
      int4 bhi = *(const int4*)(col + 32 + 16 * hi);
      v8i b;
      b[0] = blo.x; b[1] = blo.y; b[2] = blo.z; b[3] = blo.w;
      b[4] = bhi.x; b[5] = bhi.y; b[6] = bhi.z; b[7] = bhi.w;
      acc[nt] = __builtin_amdgcn_wmma_i32_16x16x64_iu8(
          true, a, true, b, acc[nt], false, false);
    }
  }

  const size_t obase = (size_t)nh * IMG_W;
  #pragma unroll
  for (int nt = 0; nt < 4; ++nt) {
    const int co = nt * 16 + m;
    #pragma unroll
    for (int r = 0; r < 8; ++r) {
      const int ow = wv * 16 + r + 8 * hi;
      __builtin_nontemporal_store((float)acc[nt][r],
                                  &out[(obase + ow) * COUT + co]);
    }
  }
}

extern "C" void kernel_launch(void* const* d_in, const int* in_sizes, int n_in,
                              void* d_out, int out_size, void* d_ws, size_t ws_size,
                              hipStream_t stream) {
  const float* x = (const float*)d_in[0];
  const float* w = (const float*)d_in[1];
  float* out = (float*)d_out;
  signed char* wb = (signed char*)d_ws;

  binarize_weights<<<(9 * 64 * 64 + 255) / 256, 256, 0, stream>>>(w, wb);
  bconv_wmma<<<64 * IMG_H, 256, 0, stream>>>(x, wb, out);
}
